// MultiheadAttention_49890340110979
// MI455X (gfx1250) — matrix operand order repaired and checked by reading, not compiled, not run on hardware
//
#include <hip/hip_runtime.h>

typedef __attribute__((ext_vector_type(16))) __bf16   bf16x16;
typedef __attribute__((ext_vector_type(8)))  float    f32x8;
typedef __attribute__((ext_vector_type(4)))  unsigned u32x4;
typedef __attribute__((ext_vector_type(4)))  unsigned uint32x4;
typedef __attribute__((ext_vector_type(8)))  int      int32x8;
typedef __attribute__((ext_vector_type(4)))  int      int32x4;

union Frag16 { u32x4 u[2]; bf16x16 v; };

#define T_LEN 512
#define B_SZ  8
#define EMB   1024
#define NH    16
#define HD    64
#define LX    512
#define S_LEN 1024
#define NROW  4096  // T*B

#if __has_builtin(__builtin_amdgcn_tensor_load_to_lds) && __has_builtin(__builtin_amdgcn_s_wait_tensorcnt)
#define HAVE_TDM 1
#else
#define HAVE_TDM 0
#endif

__device__ __forceinline__ unsigned short f2bf(float x) {
  unsigned u = __float_as_uint(x);
  u += 0x7FFFu + ((u >> 16) & 1u);   // round-to-nearest-even
  return (unsigned short)(u >> 16);
}

__device__ __forceinline__ f32x8 wmma_bf16(bf16x16 a, bf16x16 b, f32x8 c) {
  return __builtin_amdgcn_wmma_f32_16x16x32_bf16(false, a, false, b, (short)0, c, false, false);
}

#if HAVE_TDM
// 2D TDM load: tile (tile_d1 rows x tile_d0 elems) of a (tensor_d1 x tensor_d0) tensor,
// data_size = 8B. D# packing per CDNA5 ISA.
__device__ __forceinline__ void tdm_load_2d(unsigned lds_addr, unsigned long long gaddr,
                                            unsigned tensor_d0, unsigned tensor_d1,
                                            unsigned tile_d0, unsigned tile_d1,
                                            unsigned long long stride0) {
  uint32x4 g0;
  g0.x = 1u;                                               // count=1, user descriptor
  g0.y = lds_addr;                                         // LDS byte address
  g0.z = (unsigned)(gaddr & 0xFFFFFFFFull);                // global_addr[31:0]
  g0.w = (unsigned)((gaddr >> 32) & 0x01FFFFFFull) | (2u << 30);  // [56:32] | type=2
  int32x8 g1;
  g1[0] = (int)(3u << 16);                                 // data_size = 8B, no multicast
  g1[1] = (int)((tensor_d0 & 0xFFFFu) << 16);              // tensor_dim0[15:0]
  g1[2] = (int)(((tensor_d0 >> 16) & 0xFFFFu) | ((tensor_d1 & 0xFFFFu) << 16));
  g1[3] = (int)(((tensor_d1 >> 16) & 0xFFFFu) | ((tile_d0 & 0xFFFFu) << 16));
  g1[4] = (int)(tile_d1 & 0xFFFFu);                        // tile_dim1, tile_dim2=0
  g1[5] = (int)(unsigned)(stride0 & 0xFFFFFFFFull);        // tensor_dim0_stride lo
  g1[6] = (int)((stride0 >> 32) & 0xFFFFull);              // stride hi | dim1_stride=0
  g1[7] = 0;
  int32x4 gz = {0, 0, 0, 0};
#if defined(__clang_major__) && (__clang_major__ >= 23)
  int32x8 gz8 = {0, 0, 0, 0, 0, 0, 0, 0};
  __builtin_amdgcn_tensor_load_to_lds(g0, g1, gz, gz, gz8, 0);
#else
  __builtin_amdgcn_tensor_load_to_lds(g0, g1, gz, gz, 0);
#endif
}
#endif

// ---------------- fp32 -> bf16 conversion ----------------
__global__ void cvt_f32_bf16(const float* __restrict__ src,
                             unsigned short* __restrict__ dst, int n) {
  int i = blockIdx.x * blockDim.x + threadIdx.x;
  if (i < n) dst[i] = f2bf(src[i]);
}

__device__ __forceinline__ void fill_b_tile(unsigned short* dst,
                                            const unsigned short* W,
                                            int n0, int k0) {
  int tid = threadIdx.x;            // 128 threads fill 64 rows x 32 elems
  int row = tid >> 1, half = tid & 1;
  const unsigned short* src = W + (size_t)(n0 + row) * EMB + k0 + half * 16;
  *(u32x4*)(dst + row * 32 + half * 16)     = *(const u32x4*)(src);
  *(u32x4*)(dst + row * 32 + half * 16 + 8) = *(const u32x4*)(src + 8);
}

// ---------------- bf16 GEMM: C[m,n] = sum_k X[m,k] * W[n,k] ----------------
// Block = 4 waves = 128x64 C tile. B tile (32x64) staged in LDS via TDM, double-buffered.
// mode 0: (acc+bq)*0.125 -> bf16 q[bh][t][d]     mode 1: acc -> bf16 k[bh][s][d]
// mode 2: acc+bv -> bf16 vT[bh][d][s]            mode 3: acc+bo -> f32 out[row][n]
__global__ __launch_bounds__(128) void gemm_bf16(
    const unsigned short* __restrict__ X,
    const unsigned short* __restrict__ W,
    const float* __restrict__ bias,
    void* __restrict__ outp, int mode, float scale)
{
  __shared__ __align__(16) unsigned short Bl[2][64 * 32];  // [buf][n*32 + k]
  int lane = threadIdx.x & 31;
  int llo  = lane & 15, lhi = lane >> 4;
  int wv   = threadIdx.x >> 5;
  int n0   = (blockIdx.x & 15) * 64;
  int mw   = (blockIdx.x >> 4) * 128 + wv * 32;

  f32x8 acc[2][4];
#pragma unroll
  for (int i = 0; i < 2; ++i)
#pragma unroll
    for (int j = 0; j < 4; ++j) acc[i][j] = (f32x8){0,0,0,0,0,0,0,0};

  const unsigned short* xrow0 = X + (size_t)(mw + llo) * EMB;
  const unsigned short* xrow1 = X + (size_t)(mw + 16 + llo) * EMB;

#if HAVE_TDM
  unsigned lds_base = (unsigned)(size_t)&Bl[0][0];
  unsigned long long wbase =
      (unsigned long long)(size_t)W + (unsigned long long)n0 * (EMB * 2);
  int wu = __builtin_amdgcn_readfirstlane(threadIdx.x >> 5);
  if (wu == 0) {
    tdm_load_2d(lds_base, wbase, EMB / 4, EMB, 8, 64, EMB / 4);
    __builtin_amdgcn_s_wait_tensorcnt(0);
  }
  __syncthreads();
#else
  fill_b_tile(&Bl[0][0], W, n0, 0);
  __syncthreads();
#endif

  for (int k0 = 0; k0 < EMB; k0 += 32) {
    int buf = (k0 >> 5) & 1;
#if HAVE_TDM
    if (k0 + 32 < EMB) {
      int wu2 = __builtin_amdgcn_readfirstlane(threadIdx.x >> 5);
      if (wu2 == 0)
        tdm_load_2d(lds_base + (unsigned)((buf ^ 1) * 4096),
                    wbase + (unsigned long long)(k0 + 32) * 2,
                    EMB / 4, EMB, 8, 64, EMB / 4);
    }
#endif
    Frag16 a0, a1;
    a0.u[0] = *(const u32x4*)(xrow0 + k0 + lhi * 8);
    a0.u[1] = *(const u32x4*)(xrow0 + k0 + 16 + lhi * 8);
    a1.u[0] = *(const u32x4*)(xrow1 + k0 + lhi * 8);
    a1.u[1] = *(const u32x4*)(xrow1 + k0 + 16 + lhi * 8);
#pragma unroll
    for (int nt = 0; nt < 4; ++nt) {
      const unsigned short* bl = &Bl[buf][(nt * 16 + llo) * 32 + lhi * 8];
      Frag16 b;
      b.u[0] = *(const u32x4*)(bl);
      b.u[1] = *(const u32x4*)(bl + 16);
      acc[0][nt] = wmma_bf16(a0.v, b.v, acc[0][nt]);
      acc[1][nt] = wmma_bf16(a1.v, b.v, acc[1][nt]);
    }
    if (k0 + 32 < EMB) {
#if HAVE_TDM
      int wu3 = __builtin_amdgcn_readfirstlane(threadIdx.x >> 5);
      if (wu3 == 0) __builtin_amdgcn_s_wait_tensorcnt(0);
      __syncthreads();
#else
      __syncthreads();
      fill_b_tile(&Bl[buf ^ 1][0], W, n0, k0 + 32);
      __syncthreads();
#endif
    }
  }

  float bv[4];
#pragma unroll
  for (int nt = 0; nt < 4; ++nt)
    bv[nt] = bias ? bias[n0 + nt * 16 + llo] : 0.0f;

#pragma unroll
  for (int mt = 0; mt < 2; ++mt) {
#pragma unroll
    for (int nt = 0; nt < 4; ++nt) {
      int col = n0 + nt * 16 + llo;
#pragma unroll
      for (int r = 0; r < 8; ++r) {
        int row = mw + mt * 16 + r + 8 * lhi;
        float val = (acc[mt][nt][r] + bv[nt]) * scale;
        if (mode == 3) {
          ((float*)outp)[(size_t)row * EMB + col] = val;
        } else {
          int t = row >> 3, b = row & 7;  // X rows are (t*B + b)
          int h = col >> 6, dd = col & 63;
          int bh = b * NH + h;
          unsigned short* o = (unsigned short*)outp;
          if (mode == 0)      o[((size_t)bh * T_LEN + t) * HD + dd] = f2bf(val);
          else if (mode == 1) o[((size_t)bh * S_LEN + t) * HD + dd] = f2bf(val);
          else                o[((size_t)bh * HD + dd) * S_LEN + t] = f2bf(val);
        }
      }
    }
  }
}

// ---------------- expand gather: fill s in [T, T+LX) of k / vT ----------------
__global__ void gather_kernel(const int* __restrict__ idx,
                              unsigned short* __restrict__ kh,
                              unsigned short* __restrict__ vT) {
  int tid = blockIdx.x * blockDim.x + threadIdx.x;
  if (tid >= B_SZ * LX * EMB) return;
  int e = tid & (EMB - 1);
  int j = (tid >> 10) & (LX - 1);
  int b = tid >> 19;
  int h = e >> 6, dd = e & 63;
  int src = idx[b * LX + j];
  int bh = b * NH + h;
  kh[((size_t)bh * S_LEN + T_LEN + j) * HD + dd] =
      kh[((size_t)bh * S_LEN + src) * HD + dd];
  vT[((size_t)bh * HD + dd) * S_LEN + T_LEN + j] =
      vT[((size_t)bh * HD + dd) * S_LEN + src];
}

// ---------------- flash attention: per wave 16 query rows of one head ----------------
__global__ __launch_bounds__(128) void attn_kernel(
    const unsigned short* __restrict__ qh,
    const unsigned short* __restrict__ kh,
    const unsigned short* __restrict__ vT,
    const float* __restrict__ abias,
    const float* __restrict__ lw,
    unsigned short* __restrict__ attnb)
{
  __shared__ __align__(16) unsigned short plds[4][16 * 32];
  int lane  = threadIdx.x & 31;
  int llo   = lane & 15, lhi = lane >> 4;
  int wslot = threadIdx.x >> 5;
  int wid   = (blockIdx.x * blockDim.x + threadIdx.x) >> 5;
  int bh = wid >> 5;
  int t0 = (wid & 31) * 16;
  int b = bh >> 4;
  int h = bh & 15;

  Frag16 qf[2];
  {
    const unsigned short* qrow = qh + ((size_t)bh * T_LEN + t0 + llo) * HD;
#pragma unroll
    for (int kk = 0; kk < 2; ++kk) {
      qf[kk].u[0] = *(const u32x4*)(qrow + kk * 32 + lhi * 8);
      qf[kk].u[1] = *(const u32x4*)(qrow + kk * 32 + 16 + lhi * 8);
    }
  }

  f32x8 acc[4];
#pragma unroll
  for (int i = 0; i < 4; ++i) acc[i] = (f32x8){0,0,0,0,0,0,0,0};
  float mi[8], li[8];
#pragma unroll
  for (int r = 0; r < 8; ++r) { mi[r] = -3.0e38f; li[r] = 0.0f; }

  auto loadK = [&](Frag16 (&Kf)[2][2], int s0) {
#pragma unroll
    for (int hf = 0; hf < 2; ++hf) {
      const unsigned short* krow =
          kh + ((size_t)bh * S_LEN + s0 + hf * 16 + llo) * HD;
#pragma unroll
      for (int kk = 0; kk < 2; ++kk) {
        Kf[hf][kk].u[0] = *(const u32x4*)(krow + kk * 32 + lhi * 8);
        Kf[hf][kk].u[1] = *(const u32x4*)(krow + kk * 32 + 16 + lhi * 8);
      }
    }
  };

  auto process = [&](int s0, Frag16 (&Kf)[2][2]) {
    // prefetch next chunk of the bias / gate streams (global_prefetch_b8)
    if (s0 + 64 < S_LEN) {
      __builtin_prefetch(&abias[((size_t)bh * T_LEN + t0 + lhi * 8) * S_LEN + s0 + 64], 0, 1);
      __builtin_prefetch(&lw[((size_t)b * T_LEN + t0 + lhi * 8) * S_LEN + s0 + 64], 0, 1);
    }
    float w[2][8], lv[2][8];
#pragma unroll
    for (int hf = 0; hf < 2; ++hf) {
      int sb = s0 + hf * 16;
      f32x8 st = (f32x8){0,0,0,0,0,0,0,0};
#pragma unroll
      for (int kk = 0; kk < 2; ++kk)
        st = wmma_bf16(qf[kk].v, Kf[hf][kk].v, st);
#pragma unroll
      for (int r = 0; r < 8; ++r) {
        int t = t0 + r + 8 * lhi;
        int s = sb + llo;
        float bi = abias[((size_t)bh * T_LEN + t) * S_LEN + s];
        float l  = lw[((size_t)b * T_LEN + t) * S_LEN + s];
        lv[hf][r] = l;
        w[hf][r] = (st[r] + bi) * l;
      }
    }
#pragma unroll
    for (int r = 0; r < 8; ++r) {
      float tm = fmaxf(w[0][r], w[1][r]);
      tm = fmaxf(tm, __shfl_xor(tm, 1, 16));
      tm = fmaxf(tm, __shfl_xor(tm, 2, 16));
      tm = fmaxf(tm, __shfl_xor(tm, 4, 16));
      tm = fmaxf(tm, __shfl_xor(tm, 8, 16));
      float mn = fmaxf(mi[r], tm);
      float f  = __expf(mi[r] - mn);
      mi[r] = mn;
      li[r] *= f;
#pragma unroll
      for (int nt = 0; nt < 4; ++nt) acc[nt][r] *= f;
      float p0 = __expf(w[0][r] - mn);
      float p1 = __expf(w[1][r] - mn);
      li[r] += p0 + p1;
      int m = r + 8 * lhi;
      plds[wslot][m * 32 + llo]      = f2bf(p0 * lv[0][r]);
      plds[wslot][m * 32 + 16 + llo] = f2bf(p1 * lv[1][r]);
    }
    Frag16 pf;
    pf.u[0] = *(const u32x4*)&plds[wslot][llo * 32 + lhi * 8];
    pf.u[1] = *(const u32x4*)&plds[wslot][llo * 32 + 16 + lhi * 8];
#pragma unroll
    for (int nt = 0; nt < 4; ++nt) {
      const unsigned short* vrow =
          vT + ((size_t)bh * HD + nt * 16 + llo) * S_LEN + s0 + lhi * 8;
      Frag16 bfr;
      bfr.u[0] = *(const u32x4*)(vrow);
      bfr.u[1] = *(const u32x4*)(vrow + 16);
      acc[nt] = wmma_bf16(pf.v, bfr.v, acc[nt]);
    }
  };

  // software pipeline over S: two K-fragment register sets, unroll by 2
  Frag16 KA[2][2], KB[2][2];
  loadK(KA, 0);
  for (int s0 = 0; s0 < S_LEN; s0 += 64) {
    loadK(KB, s0 + 32);
    process(s0, KA);
    if (s0 + 64 < S_LEN) loadK(KA, s0 + 64);
    process(s0 + 32, KB);
  }

#pragma unroll
  for (int r = 0; r < 8; ++r) {
    float s = li[r];
    s += __shfl_xor(s, 1, 16);
    s += __shfl_xor(s, 2, 16);
    s += __shfl_xor(s, 4, 16);
    s += __shfl_xor(s, 8, 16);
    float inv = 1.0f / s;
    int t = t0 + r + 8 * lhi;
#pragma unroll
    for (int nt = 0; nt < 4; ++nt) {
      int dd = nt * 16 + llo;
      attnb[((size_t)t * B_SZ + b) * EMB + h * HD + dd] = f2bf(acc[nt][r] * inv);
    }
  }
}

extern "C" void kernel_launch(void* const* d_in, const int* in_sizes, int n_in,
                              void* d_out, int out_size, void* d_ws, size_t ws_size,
                              hipStream_t stream) {
  (void)in_sizes; (void)n_in; (void)out_size; (void)ws_size;
  const float* query = (const float*)d_in[0];
  const float* Wq    = (const float*)d_in[1];
  const float* bq    = (const float*)d_in[2];
  const float* Wk    = (const float*)d_in[3];
  const float* Wv    = (const float*)d_in[4];
  const float* bvec  = (const float*)d_in[5];
  const float* Wo    = (const float*)d_in[6];
  const float* bo    = (const float*)d_in[7];
  const float* abias = (const float*)d_in[8];
  const float* lw    = (const float*)d_in[9];
  const int*   idx   = (const int*)d_in[10];
  float* out = (float*)d_out;

  char* ws = (char*)d_ws;
  unsigned short* Xb  = (unsigned short*)(ws + (size_t)0);
  unsigned short* Wqb = (unsigned short*)(ws + ((size_t)8  << 20));
  unsigned short* Wkb = (unsigned short*)(ws + ((size_t)10 << 20));
  unsigned short* Wvb = (unsigned short*)(ws + ((size_t)12 << 20));
  unsigned short* Wob = (unsigned short*)(ws + ((size_t)14 << 20));
  unsigned short* qhb = (unsigned short*)(ws + ((size_t)16 << 20));
  unsigned short* khb = (unsigned short*)(ws + ((size_t)24 << 20));
  unsigned short* vTb = (unsigned short*)(ws + ((size_t)40 << 20));
  unsigned short* atb = (unsigned short*)(ws + ((size_t)56 << 20));

  cvt_f32_bf16<<<NROW * EMB / 256, 256, 0, stream>>>(query, Xb, NROW * EMB);
  cvt_f32_bf16<<<EMB * EMB / 256, 256, 0, stream>>>(Wq, Wqb, EMB * EMB);
  cvt_f32_bf16<<<EMB * EMB / 256, 256, 0, stream>>>(Wk, Wkb, EMB * EMB);
  cvt_f32_bf16<<<EMB * EMB / 256, 256, 0, stream>>>(Wv, Wvb, EMB * EMB);
  cvt_f32_bf16<<<EMB * EMB / 256, 256, 0, stream>>>(Wo, Wob, EMB * EMB);

  gemm_bf16<<<512, 128, 0, stream>>>(Xb, Wqb, bq,      qhb, 0, 0.125f);
  gemm_bf16<<<512, 128, 0, stream>>>(Xb, Wkb, nullptr, khb, 1, 1.0f);
  gemm_bf16<<<512, 128, 0, stream>>>(Xb, Wvb, bvec,    vTb, 2, 1.0f);

  gather_kernel<<<(B_SZ * LX * EMB) / 256, 256, 0, stream>>>(idx, khb, vTb);

  attn_kernel<<<1024, 128, 0, stream>>>(qhb, khb, vTb, abias, lw, atb);

  gemm_bf16<<<512, 128, 0, stream>>>(atb, Wob, bo, out, 3, 1.0f);
}
